// STN1D_81243601371162
// MI455X (gfx1250) — matrix operand order repaired and checked by reading, not compiled, not run on hardware
//
#include <hip/hip_runtime.h>

// ---------------------------------------------------------------------------
// CDNA5 (gfx1250, wave32) STN-1D pipeline, all GEMM-shaped work on
// v_wmma_f32_16x16x32_f16 (f16 inputs, f32 accumulate).
// ---------------------------------------------------------------------------

typedef _Float16 v8h  __attribute__((ext_vector_type(8)));
typedef _Float16 v16h __attribute__((ext_vector_type(16)));
typedef float    v8f  __attribute__((ext_vector_type(8)));

static constexpr int KSPLIT = 16;     // K-split factor for dense1
static constexpr int KD     = 32768;  // dense1 reduction depth (1024*32)

// Build a v16h fragment from two 16-byte-aligned groups of 8 halfs.
// Matches the CDNA5 16-bit A/B operand layout: element i -> VGPR i/2 half i%2.
__device__ __forceinline__ v16h pack16(const _Float16* p0, const _Float16* p1) {
    v8h lo = *(const v8h*)p0;
    v8h hi = *(const v8h*)p1;
    v16h r;
#pragma unroll
    for (int i = 0; i < 8; ++i) { r[i] = lo[i]; r[8 + i] = hi[i]; }
    return r;
}

// ---------------------------------------------------------------------------
// Fused conv1d(k=3, SAME) + bias + ReLU + maxpool2 via implicit-GEMM WMMA.
//   A: 16 positions x K (K = 3*CIN, im2col rows read from padded row-major
//      LDS because the 3-tap window concatenation == 3 consecutive rows)
//   B: K x 32 output channels (weights staged column-major in LDS)
// One workgroup = 4 waves = 64 pre-pool positions of one batch row.
// ---------------------------------------------------------------------------
template <int CIN_, typename TIN>
__global__ __launch_bounds__(128, 2) void conv_relu_pool_wmma(
    const TIN* __restrict__ xin, const float* __restrict__ wgt,
    const float* __restrict__ bias, _Float16* __restrict__ out, int Lin)
{
    constexpr int COUT_ = 32;
    constexpr int KTOT  = 3 * CIN_;      // im2col depth (192 or 96)
    constexpr int TILE  = 64;            // pre-pool positions per workgroup
    constexpr int RS    = CIN_ + 8;      // padded LDS row stride (halfs) -> bank spread

    __shared__ _Float16 lx[(TILE + 2) * RS];   // input rows t0-1 .. t0+TILE
    __shared__ _Float16 lw[COUT_ * KTOT];      // weights, column-major per out-channel

    const int tilesPerRow = Lin / TILE;
    const int b   = blockIdx.x / tilesPerRow;
    const int t0  = (blockIdx.x % tilesPerRow) * TILE;
    const int tid = threadIdx.x;

    // Stage weights: global layout (k, ci, co) -> lw[co*KTOT + k*CIN_+ci]
    for (int i = tid; i < KTOT * COUT_; i += blockDim.x) {
        int co = i % COUT_, kci = i / COUT_;
        lw[co * KTOT + kci] = (_Float16)wgt[i];
    }
    // Stage input tile (+halo), zero-padded at sequence edges, f32 -> f16
    for (int i = tid; i < (TILE + 2) * CIN_; i += blockDim.x) {
        int r = i / CIN_, c = i % CIN_;
        int pos = t0 - 1 + r;
        float v = (pos >= 0 && pos < Lin) ? (float)xin[((long)b * Lin + pos) * CIN_ + c] : 0.0f;
        lx[r * RS + c] = (_Float16)v;
    }
    __syncthreads();

    const int lane = tid & 31;
    const int wv   = tid >> 5;           // 4 waves x 16 M-rows
    const int m0   = wv * 16;
    const int mr   = m0 + (lane & 15);   // this lane's A row (== lds halo row base)
    const int sel  = lane >> 4;          // which K-half this lane holds
    const int ch   = lane & 15;

    v8f acc0 = {}; v8f acc1 = {};
#pragma unroll
    for (int kc = 0; kc < KTOT / 32; ++kc) {
        // A fragment: lane<16 -> K 0..15, lane>=16 -> K 16..31
        const int g0 = kc * 32 + sel * 16;
        const int g1 = g0 + 8;
        v16h a = pack16(&lx[(mr + g0 / CIN_) * RS + (g0 % CIN_)],
                        &lx[(mr + g1 / CIN_) * RS + (g1 % CIN_)]);
        // B fragments: lane<16 -> K 0..15 of col n, lane>=16 -> K 16..31
        const int kb = kc * 32 + sel * 16;
        v16h bm0 = pack16(&lw[ch * KTOT + kb],        &lw[ch * KTOT + kb + 8]);
        v16h bm1 = pack16(&lw[(16 + ch) * KTOT + kb], &lw[(16 + ch) * KTOT + kb + 8]);
        acc0 = __builtin_amdgcn_wmma_f32_16x16x32_f16(false, a, false, bm0, (short)0, acc0, false, false);
        acc1 = __builtin_amdgcn_wmma_f32_16x16x32_f16(false, a, false, bm1, (short)0, acc1, false, false);
    }

    // bias + ReLU + pool2 straight from the C fragment:
    // VGPR r holds M=r (lanes 0-15) / M=r+8 (lanes 16-31); pool pairs (2i,2i+1).
    const float b0 = bias[ch], b1 = bias[16 + ch];
    const int Lout  = Lin >> 1;
    const int pbase = (t0 + m0) >> 1;
    const int prow  = (lane < 16) ? 0 : 4;
#pragma unroll
    for (int i = 0; i < 4; ++i) {
        float p0 = fmaxf(fmaxf(acc0[2 * i], acc0[2 * i + 1]) + b0, 0.0f);
        float p1 = fmaxf(fmaxf(acc1[2 * i], acc1[2 * i + 1]) + b1, 0.0f);
        long o = ((long)b * Lout + (pbase + prow + i)) * COUT_;
        out[o + ch]      = (_Float16)p0;
        out[o + 16 + ch] = (_Float16)p1;
    }
}

// ---------------------------------------------------------------------------
// dense1: (128 x 32768) @ (32768 x 32), K-split, deterministic partials
// (no float atomics -> bitwise-identical across replays).
// One wave per workgroup; grid = mtile(8) x ntile(2) x kslice(KSPLIT).
// ---------------------------------------------------------------------------
__global__ __launch_bounds__(32) void dense1_wmma(
    const _Float16* __restrict__ h2, const float* __restrict__ wd1,
    float* __restrict__ part)
{
    const int ks   = blockIdx.x % KSPLIT;
    const int nt   = (blockIdx.x / KSPLIT) & 1;
    const int mt   = blockIdx.x / (KSPLIT * 2);
    const int lane = threadIdx.x & 31;
    const int sel  = lane >> 4;
    const int ch   = lane & 15;
    const int m    = mt * 16 + ch;       // A row (batch index)
    const int n    = nt * 16 + ch;       // B column (output channel)

    const int chunk = KD / KSPLIT;       // 2048 K per slice
    v8f acc = {};
    for (int kc = 0; kc < chunk / 32; ++kc) {
        const int kb = ks * chunk + kc * 32;
        // prefetch next A chunk (global_prefetch_b8)
        __builtin_prefetch(h2 + (long)m * KD + kb + 32, 0, 0);
        const _Float16* pa = h2 + (long)m * KD + kb + sel * 16;
        v16h a = pack16(pa, pa + 8);
        const int k0 = kb + sel * 16;
        v16h bm;
#pragma unroll
        for (int i = 0; i < 16; ++i) bm[i] = (_Float16)wd1[(long)(k0 + i) * 32 + n];
        acc = __builtin_amdgcn_wmma_f32_16x16x32_f16(false, a, false, bm, (short)0, acc, false, false);
    }
    float* dst = part + (long)ks * 128 * 32;   // disjoint slot per (mt,nt,ks)
#pragma unroll
    for (int r = 0; r < 8; ++r) {
        int row = mt * 16 + r + ((lane < 16) ? 0 : 8);
        dst[row * 32 + nt * 16 + ch] = acc[r];
    }
}

// ---------------------------------------------------------------------------
// Reduce K-split partials in fixed order, bias+ReLU, dense2 -> (scale, shift)
// ---------------------------------------------------------------------------
__global__ __launch_bounds__(128) void dense_head(
    const float* __restrict__ part, const float* __restrict__ bd1,
    const float* __restrict__ wd2, const float* __restrict__ bd2,
    float* __restrict__ ss)
{
    const int b = threadIdx.x;           // one block of 128 threads, b = batch
    float p0 = bd2[0], p1 = bd2[1];
    for (int c = 0; c < 32; ++c) {
        float s = 0.0f;
        for (int ks = 0; ks < KSPLIT; ++ks) s += part[((long)ks * 128 + b) * 32 + c];
        float h = fmaxf(s + bd1[c], 0.0f);
        p0 += h * wd2[c * 2 + 0];
        p1 += h * wd2[c * 2 + 1];
    }
    ss[b * 2 + 0] = p0 + 1.0f;           // scale = params[:,0] + 1
    ss[b * 2 + 1] = p1;                  // shift = params[:,1]
}

// ---------------------------------------------------------------------------
// Warp + linear resample of original x. 16 lanes per (b,l), float4 per lane.
// Streams ~256 MB in / 128 MB out -> the HBM-bound tail of the pipeline.
// ---------------------------------------------------------------------------
__global__ __launch_bounds__(256) void warp_sample(
    const float* __restrict__ x, const float* __restrict__ ss,
    float* __restrict__ out, int L)
{
    const long t  = (long)blockIdx.x * 256 + threadIdx.x;   // B*L*16 threads
    const int cg  = (int)(t & 15);                          // 4-channel group
    const long bl = t >> 4;
    const int l   = (int)(bl % L);
    const int b   = (int)(bl / L);

    const float scale = ss[b * 2 + 0];
    const float shift = ss[b * 2 + 1];
    const float src = -1.0f + 2.0f * (float)l / (float)(L - 1);
    float tg = fminf(fmaxf(scale * src + shift, -1.0f), 1.0f);
    float ti = (tg + 1.0f) * 0.5f * (float)(L - 1);
    float f0 = floorf(ti);
    int i0 = (int)f0;
    int i1 = i0 + 1; if (i1 > L - 1) i1 = L - 1;
    float w1 = ti - f0, w0 = 1.0f - w1;

    const float4 v0 = ((const float4*)(x + ((long)b * L + i0) * 64))[cg];
    const float4 v1 = ((const float4*)(x + ((long)b * L + i1) * 64))[cg];
    float4 o;
    o.x = w0 * v0.x + w1 * v1.x;
    o.y = w0 * v0.y + w1 * v1.y;
    o.z = w0 * v0.z + w1 * v1.z;
    o.w = w0 * v0.w + w1 * v1.w;
    ((float4*)(out + ((long)b * L + l) * 64))[cg] = o;
}

// ---------------------------------------------------------------------------
extern "C" void kernel_launch(void* const* d_in, const int* in_sizes, int n_in,
                              void* d_out, int out_size, void* d_ws, size_t ws_size,
                              hipStream_t stream)
{
    (void)in_sizes; (void)n_in; (void)out_size;
    const int B = 128, L = 4096;

    const float* x   = (const float*)d_in[0];
    const float* wc1 = (const float*)d_in[1];
    const float* bc1 = (const float*)d_in[2];
    const float* wc2 = (const float*)d_in[3];
    const float* bc2 = (const float*)d_in[4];
    const float* wd1 = (const float*)d_in[5];
    const float* bd1 = (const float*)d_in[6];
    const float* wd2 = (const float*)d_in[7];
    const float* bd2 = (const float*)d_in[8];
    float* out = (float*)d_out;

    // Small scratch MUST live in d_ws: warp_sample reads `ss` while it
    // overwrites all of d_out, so these may not alias the output buffer.
    float* ss   = (float*)d_ws;                               // 128*2 f32
    float* part = (float*)((char*)d_ws + 4096);               // KSPLIT*128*32 f32
    size_t small_bytes = 4096 + (size_t)KSPLIT * 128 * 32 * sizeof(float);

    // Big f16 activations: prefer d_ws; else stage in d_out (fully consumed
    // before warp_sample launches, then completely overwritten by it).
    const size_t h1_bytes = (size_t)B * (L / 2) * 32 * sizeof(_Float16);  // 16 MB
    const size_t h2_bytes = (size_t)B * (L / 4) * 32 * sizeof(_Float16);  //  8 MB
    size_t big_off = (small_bytes + 255) & ~(size_t)255;
    _Float16 *h1, *h2;
    if (ws_size >= big_off + h1_bytes + h2_bytes) {
        h1 = (_Float16*)((char*)d_ws + big_off);
        h2 = (_Float16*)((char*)h1 + h1_bytes);
    } else {
        h1 = (_Float16*)d_out;
        h2 = (_Float16*)((char*)d_out + h1_bytes);
    }

    conv_relu_pool_wmma<64, float><<<B * (L / 64), 128, 0, stream>>>(x, wc1, bc1, h1, L);
    conv_relu_pool_wmma<32, _Float16><<<B * ((L / 2) / 64), 128, 0, stream>>>(h1, wc2, bc2, h2, L / 2);
    dense1_wmma<<<8 * 2 * KSPLIT, 32, 0, stream>>>(h2, wd1, part);
    dense_head<<<1, 128, 0, stream>>>(part, bd1, wd2, bd2, ss);
    warp_sample<<<(int)(((long)B * L * 16) / 256), 256, 0, stream>>>(x, ss, out, L);
}
